// StitchEncoder_75995151335989
// MI455X (gfx1250) — matrix operand order repaired and checked by reading, not compiled, not run on hardware
//
#include <hip/hip_runtime.h>

// ---------------------------------------------------------------------------
// StitchEncoder: grouped (per-expert) Linear(512,1024) -> Softsign ->
// Linear(1024,512), dispatched by eid, fused into one gfx1250 WMMA kernel.
//
// Precision: bf16 operands with f32 accumulation via v_wmma_f32_16x16x32_bf16.
// ---------------------------------------------------------------------------

typedef __bf16 bf16;
typedef __attribute__((ext_vector_type(16))) __bf16 v16bf;
typedef __attribute__((ext_vector_type(8)))  __bf16 bf16x8;
typedef __attribute__((ext_vector_type(4)))  __bf16 bf16x4;
typedef __attribute__((ext_vector_type(2)))  __bf16 bf16x2;
typedef __attribute__((ext_vector_type(8)))  float  v8f;
typedef __attribute__((ext_vector_type(4)))  float  f32x4;

#define B_   128
#define F_   100
#define N_   512
#define H_   1024
#define P_   512

#define MROWS 64          // F rows handled per workgroup (chunk)
#define MT    4           // 4 M-tiles of 16 rows
#define KB    32          // K depth per WMMA / per staged weight slab

// LDS strides in bf16 elements. All are multiples of 8 so every 16B vector
// load is aligned; chosen so dword row-stride % 64 spreads banks:
//  520/2=260 dw, 260%64=4  -> A-operand lanes hit disjoint bank quads
//  1032/2=516 dw, 516%64=4 -> same for the act tile
//  col stride 40*2/4=20 dw -> B-operand lanes 0..15 hit distinct bank quads,
//                             and staging dword stores are only 2-way conflicted
#define XS_STRIDE  520
#define ACT_STRIDE 1032
#define WT_STRIDE  40

#define XS_BYTES   (MROWS * XS_STRIDE * 2)          // 66,560
#define ACT_BYTES  (MROWS * ACT_STRIDE * 2)         // 132,096
#define WT_BYTES   (512 * WT_STRIDE * 2)            // 40,960
#define SMEM_BYTES (XS_BYTES + ACT_BYTES + 2 * WT_BYTES)   // 280,576 <= 320K

union V16 { v16bf v; bf16x8 h[2]; };

// A operand (16x32 bf16, ISA 7.12.2 16-bit A layout):
//  lanes 0-15 : M=lane,    VGPR0-3 = K 0..7,  VGPR4-7 = K 16..23
//  lanes16-31 : M=lane-16, VGPR0-3 = K 8..15, VGPR4-7 = K 24..31
__device__ __forceinline__ v16bf load_a(const bf16* __restrict__ base, int lda,
                                        int kcol, int lane) {
  const int m  = lane & 15;
  const int k0 = (lane & 16) ? 8 : 0;
  const bf16* p = base + m * lda + kcol + k0;
  V16 u;
  u.h[0] = *(const bf16x8*)(p);        // ds_load_b128
  u.h[1] = *(const bf16x8*)(p + 16);   // ds_load_b128
  return u.v;
}

// B operand (32x16 bf16), same K order as A: lanes 0-15 hold K 0..7,16..23
// of col N=lane, lanes 16-31 hold K 8..15,24..31. Weight slab is stored
// transposed in LDS: wt[col][k].
__device__ __forceinline__ v16bf load_b(const bf16* __restrict__ wt,
                                        int col0, int lane) {
  const int c  = col0 + (lane & 15);
  const int k0 = (lane & 16) ? 8 : 0;
  const bf16* p = wt + c * WT_STRIDE + k0;
  V16 u;
  u.h[0] = *(const bf16x8*)(p);        // ds_load_b128
  u.h[1] = *(const bf16x8*)(p + 16);   // ds_load_b128
  return u.v;
}

// Stage a KB x 512 f32 weight slab -> bf16, transposed, into LDS.
// Each thread owns a K-pair (k, k+1) and one column per lane step:
//   - global reads: consecutive lanes read consecutive columns (fully
//     coalesced b32 bursts, two rows)
//   - the two bf16 (k, k+1) are packed into one dword (v_cvt_pk_bf16_f32)
//     and written with a single ds_store_b32 at wt[col][k]
__device__ __forceinline__ void stage_w(bf16* __restrict__ wt,
                                        const float* __restrict__ W, int ldw,
                                        int krow0, int col0, int tid) {
  const int kp = (tid >> 5) * 2;    // 0,2,...,30 : K pair inside slab
  const int cl = tid & 31;          // column lane
  const float* src0 = W + (size_t)(krow0 + kp) * (size_t)ldw + col0;
  const float* src1 = src0 + ldw;
#pragma unroll
  for (int j = 0; j < 16; ++j) {
    const int c = cl + j * 32;
    const float lo = src0[c];
    const float hi = src1[c];
    bf16x2 p = {(bf16)lo, (bf16)hi};
    *(bf16x2*)(wt + c * WT_STRIDE + kp) = p;   // ds_store_b32
  }
}

// GEMM: A [MROWS x NKB*32] (bf16, LDS) x W [NKB*32 x 512] (f32, global,
// streamed through a double-buffered bf16 LDS slab) += acc.
// Wave w owns output column tiles {w, w+16} across all 4 M tiles.
template <int NKB>
__device__ __forceinline__ void gemm_block(const bf16* __restrict__ A, int lda,
                                           const float* __restrict__ W, int ldw,
                                           int col0,
                                           bf16* __restrict__ wt0,
                                           bf16* __restrict__ wt1,
                                           int tid, int wave, int lane,
                                           v8f (&acc)[2][MT]) {
  stage_w(wt0, W, ldw, 0, col0, tid);
  __syncthreads();
#pragma unroll 2
  for (int kb = 0; kb < NKB; ++kb) {
    bf16* cur = (kb & 1) ? wt1 : wt0;
    bf16* nxt = (kb & 1) ? wt0 : wt1;
    if (kb + 1 < NKB) stage_w(nxt, W, ldw, (kb + 1) * KB, col0, tid);

    v16bf a[MT];
#pragma unroll
    for (int mt = 0; mt < MT; ++mt)
      a[mt] = load_a(A + mt * 16 * lda, lda, kb * KB, lane);

#pragma unroll
    for (int ci = 0; ci < 2; ++ci) {
      const v16bf bm = load_b(cur, (wave + ci * 16) * 16, lane);
#pragma unroll
      for (int mt = 0; mt < MT; ++mt)
        acc[ci][mt] = __builtin_amdgcn_wmma_f32_16x16x32_bf16(
            false, a[mt], false, bm, (short)0, acc[ci][mt], false, false);
    }
    __syncthreads();   // staged slab (kb+1) ready; slab kb free for reuse
  }
}

__global__ void __launch_bounds__(512)
StitchEncoder_75995151335989_kernel(const float* __restrict__ x,
                                    const int*   __restrict__ eid,
                                    const float* __restrict__ W1,
                                    const float* __restrict__ b1,
                                    const float* __restrict__ W2,
                                    const float* __restrict__ b2,
                                    float* __restrict__ out) {
  extern __shared__ char smem[];
  bf16* xs  = (bf16*)(smem);                          // [MROWS][XS_STRIDE]
  bf16* act = (bf16*)(smem + XS_BYTES);               // [MROWS][ACT_STRIDE]
  bf16* wt0 = (bf16*)(smem + XS_BYTES + ACT_BYTES);   // [512][WT_STRIDE]
  bf16* wt1 = (bf16*)(smem + XS_BYTES + ACT_BYTES + WT_BYTES);

  const int tid   = threadIdx.x;
  const int wave  = tid >> 5;
  const int lane  = tid & 31;
  const int bidx  = blockIdx.x >> 1;   // trial
  const int chunk = blockIdx.x & 1;    // 64-row F chunk
  const int e     = eid[bidx];

  const float* W1e = W1 + (size_t)e * N_ * H_;
  const float* W2e = W2 + (size_t)e * H_ * P_;
  const float* b1e = b1 + (size_t)e * H_;
  const float* b2e = b2 + (size_t)e * P_;

  // ---- Stage x chunk, f32 -> bf16, zero-padding rows >= F_ ----------------
  {
    const int row = tid >> 3;           // 0..63
    const int cb  = (tid & 7) * 4;      // 0..28
    const int gf  = chunk * MROWS + row;
    const float* src = x + ((size_t)bidx * F_ + gf) * N_;
#pragma unroll
    for (int j = 0; j < 16; ++j) {
      const int c = cb + j * 32;
      f32x4 v = {0.f, 0.f, 0.f, 0.f};
      if (gf < F_) v = *(const f32x4*)(src + c);
      bf16x4 h = {(bf16)v.x, (bf16)v.y, (bf16)v.z, (bf16)v.w};
      *(bf16x4*)(xs + row * XS_STRIDE + c) = h;   // ds_store_b64
    }
  }
  // (first barrier inside gemm_block orders xs writes vs. A reads)

  const int rbase = (lane & 16) ? 8 : 0;   // C/D layout: lanes>=16 -> M=r+8

  // ---- Phase 1: stitched = x @ W1 + b1, softsign -> act (bf16 in LDS) ----
#pragma unroll 1
  for (int hh = 0; hh < 2; ++hh) {
    v8f acc[2][MT];
#pragma unroll
    for (int ci = 0; ci < 2; ++ci)
#pragma unroll
      for (int mt = 0; mt < MT; ++mt)
        acc[ci][mt] = (v8f){0.f, 0.f, 0.f, 0.f, 0.f, 0.f, 0.f, 0.f};

    gemm_block<N_ / KB>(xs, XS_STRIDE, W1e, H_, hh * 512, wt0, wt1,
                        tid, wave, lane, acc);

#pragma unroll
    for (int ci = 0; ci < 2; ++ci) {
      const int col  = (wave + ci * 16) * 16 + (lane & 15);
      const float bias = b1e[hh * 512 + col];
#pragma unroll
      for (int mt = 0; mt < MT; ++mt) {
#pragma unroll
        for (int r = 0; r < 8; ++r) {
          float s = acc[ci][mt][r] + bias;
          s = s / (1.0f + __builtin_fabsf(s));   // Softsign, SCALE = 1.0
          act[(mt * 16 + rbase + r) * ACT_STRIDE + hh * 512 + col] = (bf16)s;
        }
      }
    }
    __syncthreads();
  }

  // ---- Phase 2: out = act @ W2 + b2 --------------------------------------
  v8f acc2[2][MT];
#pragma unroll
  for (int ci = 0; ci < 2; ++ci)
#pragma unroll
    for (int mt = 0; mt < MT; ++mt)
      acc2[ci][mt] = (v8f){0.f, 0.f, 0.f, 0.f, 0.f, 0.f, 0.f, 0.f};

  gemm_block<H_ / KB>(act, ACT_STRIDE, W2e, P_, 0, wt0, wt1,
                      tid, wave, lane, acc2);

#pragma unroll
  for (int ci = 0; ci < 2; ++ci) {
    const int col  = (wave + ci * 16) * 16 + (lane & 15);
    const float bias = b2e[col];
#pragma unroll
    for (int mt = 0; mt < MT; ++mt) {
#pragma unroll
      for (int r = 0; r < 8; ++r) {
        const int f = chunk * MROWS + mt * 16 + rbase + r;
        if (f < F_)
          out[((size_t)bidx * F_ + f) * P_ + col] = acc2[ci][mt][r] + bias;
      }
    }
  }
}

extern "C" void kernel_launch(void* const* d_in, const int* in_sizes, int n_in,
                              void* d_out, int out_size, void* d_ws, size_t ws_size,
                              hipStream_t stream) {
  const float* x   = (const float*)d_in[0];
  const int*   eid = (const int*)  d_in[1];
  const float* W1  = (const float*)d_in[2];
  const float* b1  = (const float*)d_in[3];
  const float* W2  = (const float*)d_in[4];
  const float* b2  = (const float*)d_in[5];
  float* out = (float*)d_out;

  (void)in_sizes; (void)n_in; (void)out_size; (void)d_ws; (void)ws_size;

  hipFuncSetAttribute((const void*)StitchEncoder_75995151335989_kernel,
                      hipFuncAttributeMaxDynamicSharedMemorySize,
                      (int)SMEM_BYTES);

  StitchEncoder_75995151335989_kernel<<<dim3(B_ * 2), dim3(512), SMEM_BYTES,
                                        stream>>>(x, eid, W1, b1, W2, b2, out);
}
